// EfficientTransformerBlock_76441827934744
// MI455X (gfx1250) — matrix operand order repaired and checked by reading, not compiled, not run on hardware
//
#include <hip/hip_runtime.h>
#include <hip/hip_bf16.h>

// ---------------- problem constants ----------------
#define B_  2
#define L_  2048
#define D_  1024
#define H_  16
#define KV_ 4
#define HD_ 64
#define FF_ 4096
#define M_  (B_*L_)          // 4096 rows total
#define KVD_ (KV_*HD_)       // 256

typedef __bf16 bf16;
typedef __attribute__((ext_vector_type(16))) __bf16 v16bf;
typedef __attribute__((ext_vector_type(8)))  float  v8f;
typedef __attribute__((ext_vector_type(4)))  unsigned u32x4;
typedef __attribute__((ext_vector_type(8)))  int      i32x8;
typedef __attribute__((ext_vector_type(4)))  int      i32x4;

#if defined(__has_builtin)
#  if __has_builtin(__builtin_amdgcn_tensor_load_to_lds) && __has_builtin(__builtin_amdgcn_s_wait_tensorcnt)
#    define USE_TDM 1
#  endif
#endif
#ifndef USE_TDM
#  define USE_TDM 0
#endif

// D = A(16x32 bf16) x B(32x16 bf16) + C(16x16 f32), wave32 WMMA
static __device__ __forceinline__ v8f wmma_bf16(v16bf a, v16bf b, v8f c) {
  return __builtin_amdgcn_wmma_f32_16x16x32_bf16(false, a, false, b, (short)0, c, false, false);
}
// A-frag (16x32): lane l: row l%16, kbase=(l/16)*8; elems 0..7 = K kbase.., 8..15 = K 16+kbase..
static __device__ __forceinline__ v16bf frag_a(const bf16* p) {
  v16bf f;
#pragma unroll
  for (int i = 0; i < 8; i++) { f[i] = p[i]; f[i + 8] = p[i + 16]; }
  return f;
}
// B-frag (32x16): lane l: col l%16, kbase=(l/16)*8; elems 0..7 = K kbase.., 8..15 = K 16+kbase..
static __device__ __forceinline__ v16bf frag_b(const bf16* p) {
  v16bf f;
#pragma unroll
  for (int i = 0; i < 8; i++) { f[i] = p[i]; f[i + 8] = p[i + 16]; }
  return f;
}

#if USE_TDM
// low 32 bits of flat shared-aperture address = workgroup LDS byte offset
static __device__ __forceinline__ unsigned lds_off(const void* p) {
  return (unsigned)(unsigned long long)(uintptr_t)p;
}
// Issue a 2-D tile DMA (bf16 elements): tile_d0 elems/row, tile_d1 rows,
// row stride stride0 (elems). Optional LDS pad: after every (2<<padi) dwords
// insert (pada+1) dwords.  D# built per CDNA5 ISA §8 (groups 0/1; 2/3 zero,
// tile_dim2=0 so higher dims unused).
static __device__ __forceinline__ void tdm_load_2d(
    const bf16* gaddr, unsigned ldsoff, unsigned tile_d0, unsigned tile_d1,
    unsigned long long stride0, int pad_en, unsigned padi, unsigned pada) {
  unsigned long long ga = (unsigned long long)(uintptr_t)gaddr;
  u32x4 g0;
  g0[0] = 1u;                                             // count=1 (valid), user mode
  g0[1] = ldsoff;                                         // lds_addr
  g0[2] = (unsigned)(ga & 0xFFFFFFFFu);                   // global_addr[31:0]
  g0[3] = (unsigned)((ga >> 32) & 0x1FFFFFFu) | (2u << 30); // ga[56:32] | type=2
  i32x8 g1;
  unsigned dw0 = (1u << 16);                              // data_size=1 (2 bytes)
  if (pad_en) dw0 |= (1u << 20) | (padi << 22) | (pada << 25);
  g1[0] = (int)dw0;
  g1[1] = (int)((tile_d0 & 0xFFFFu) << 16);               // tensor_dim0 = tile_d0
  g1[2] = (int)((tile_d1 & 0xFFFFu) << 16);               // td0 hi=0 | tensor_dim1 = tile_d1
  g1[3] = (int)((tile_d0 & 0xFFFFu) << 16);               // td1 hi=0 | tile_dim0
  g1[4] = (int)(tile_d1 & 0xFFFFu);                       // tile_dim1 | tile_dim2=0
  g1[5] = (int)(unsigned)(stride0 & 0xFFFFFFFFull);       // tensor_dim0_stride[31:0]
  g1[6] = (int)(unsigned)((stride0 >> 32) & 0xFFFFull);   // stride0[47:32] | stride1 lo=0
  g1[7] = 0;
  i32x4 z4 = {0, 0, 0, 0};
#if __clang_major__ >= 23
  i32x8 z8 = {0, 0, 0, 0, 0, 0, 0, 0};
  __builtin_amdgcn_tensor_load_to_lds(g0, g1, z4, z4, z8, 0);
#else
  __builtin_amdgcn_tensor_load_to_lds(g0, g1, z4, z4, 0);
#endif
}
#endif

// ---------------- fp32 [K,N] -> bf16 transposed [N,K] ----------------
__global__ __launch_bounds__(256) void cvt_transpose_k(const float* __restrict__ in,
                                                       bf16* __restrict__ out,
                                                       int K, int N) {
  __shared__ float t[32][33];
  const int kt = blockIdx.y * 32, nt = blockIdx.x * 32;
  for (int i = threadIdx.x; i < 32 * 32; i += 256) {
    int r = i >> 5, c = i & 31;
    t[r][c] = in[(size_t)(kt + r) * N + nt + c];
  }
  __syncthreads();
  for (int i = threadIdx.x; i < 32 * 32; i += 256) {
    int r = i >> 5, c = i & 31;       // r = n index, c = k index
    out[(size_t)(nt + r) * K + kt + c] = (bf16)t[c][r];
  }
}

// ---------------- layernorm (one row / block) ----------------
__global__ __launch_bounds__(256) void layernorm_bf16_k(const float* __restrict__ x,
                                                        const float* __restrict__ g,
                                                        const float* __restrict__ bt,
                                                        bf16* __restrict__ out) {
  __shared__ float red[16];
  const int row = blockIdx.x;
  const float* xr = x + (size_t)row * D_;
  float s = 0.f, s2 = 0.f;
  for (int i = threadIdx.x; i < D_; i += 256) { float v = xr[i]; s += v; s2 += v * v; }
#pragma unroll
  for (int o = 16; o > 0; o >>= 1) { s += __shfl_xor(s, o, 32); s2 += __shfl_xor(s2, o, 32); }
  int wave = threadIdx.x >> 5, lane = threadIdx.x & 31;
  if (lane == 0) { red[wave] = s; red[8 + wave] = s2; }
  __syncthreads();
  if (threadIdx.x == 0) {
    float a = 0.f, b2 = 0.f;
    for (int w = 0; w < 8; w++) { a += red[w]; b2 += red[8 + w]; }
    red[0] = a; red[8] = b2;
  }
  __syncthreads();
  float mean = red[0] * (1.f / D_);
  float var  = red[8] * (1.f / D_) - mean * mean;
  float rstd = rsqrtf(var + 1e-5f);
  for (int i = threadIdx.x; i < D_; i += 256)
    out[(size_t)row * D_ + i] = (bf16)((xr[i] - mean) * rstd * g[i] + bt[i]);
}

// ---------------- tiled WMMA GEMM: C = A[MxK] @ W  (W given TRANSPOSED [N,K]) ----
// 128x128 macro tile, K-tile 32, 8 waves 2(M)x4(N), wave tile 64x32.
// Double-buffered LDS; tiles fetched by the Tensor Data Mover (wave 0 issues,
// waits TENSORcnt, workgroup barrier) when available.
// EPI: 0 bf16 | 1 silu->bf16 | 2 *mulm->bf16 | 3 resid+ ->f32
//      4 resid + a*C + (1-a)*mixm ->f32 (a=sigmoid(gatep[col])) | 5 bf16 transposed store
template <int EPI>
__global__ __launch_bounds__(256) void gemm_bf16_k(
    const bf16* __restrict__ A, const bf16* __restrict__ Wt,
    int M, int N, int K,
    const float* __restrict__ resid, const bf16* __restrict__ mulm,
    const bf16* __restrict__ mixm, const float* __restrict__ gatep,
    float* __restrict__ outf, bf16* __restrict__ outb) {
  __shared__ bf16 sA[2][128 * 40];    // [row][k] 32 + 8 pad halfs
  __shared__ bf16 sB[2][128 * 40];    // [n][k]   32 + 8 pad halfs
  const int tid = threadIdx.x, lane = tid & 31, wave = tid >> 5;
  const int m0 = blockIdx.y * 128, n0 = blockIdx.x * 128;
  const int wm = (wave >> 2) * 64, wn = (wave & 3) * 32;
  v8f acc[4][2];
  v8f zero = {};
#pragma unroll
  for (int i = 0; i < 4; i++)
#pragma unroll
    for (int j = 0; j < 2; j++) acc[i][j] = zero;

#if USE_TDM
  // prologue: DMA first K-tile (pad 16 dwords data + 4 dwords -> 40-half rows)
  if (wave == 0) {
    tdm_load_2d(A  + (size_t)m0 * K, lds_off(&sA[0][0]), 32, 128, (unsigned long long)K, 1, 3, 3);
    tdm_load_2d(Wt + (size_t)n0 * K, lds_off(&sB[0][0]), 32, 128, (unsigned long long)K, 1, 3, 3);
    __builtin_amdgcn_s_wait_tensorcnt(0);
  }
  __syncthreads();
#endif
  int buf = 0;
  for (int k0 = 0; k0 < K; k0 += 32, buf ^= 1) {
#if USE_TDM
    if (k0 + 32 < K && wave == 0) {   // DMA next tile while computing this one
      tdm_load_2d(A  + (size_t)m0 * K + k0 + 32, lds_off(&sA[buf ^ 1][0]), 32, 128,
                  (unsigned long long)K, 1, 3, 3);
      tdm_load_2d(Wt + (size_t)n0 * K + k0 + 32, lds_off(&sB[buf ^ 1][0]), 32, 128,
                  (unsigned long long)K, 1, 3, 3);
    }
#else
    // synchronous staging fallback: 16B chunks, both tiles row-contiguous
#pragma unroll
    for (int r = 0; r < 2; r++) {
      int idx = tid + r * 256, row = idx >> 2, seg = idx & 3;
      *(uint4*)&sA[buf][row * 40 + seg * 8] =
          *(const uint4*)(A + (size_t)(m0 + row) * K + k0 + seg * 8);
      *(uint4*)&sB[buf][row * 40 + seg * 8] =
          *(const uint4*)(Wt + (size_t)(n0 + row) * K + k0 + seg * 8);
    }
    if (k0 + 32 < K) {
      __builtin_prefetch(A + (size_t)(m0 + (tid >> 1)) * K + k0 + 32, 0, 0);
      __builtin_prefetch(Wt + (size_t)(n0 + (tid >> 1)) * K + k0 + 32, 0, 0);
    }
    __syncthreads();
#endif
    v16bf af[4], bfr[2];
#pragma unroll
    for (int i = 0; i < 4; i++)
      af[i] = frag_a(&sA[buf][(wm + i * 16 + (lane & 15)) * 40 + (lane >> 4) * 8]);
#pragma unroll
    for (int j = 0; j < 2; j++)
      bfr[j] = frag_b(&sB[buf][(wn + j * 16 + (lane & 15)) * 40 + (lane >> 4) * 8]);
#pragma unroll
    for (int i = 0; i < 4; i++)
#pragma unroll
      for (int j = 0; j < 2; j++) acc[i][j] = wmma_bf16(af[i], bfr[j], acc[i][j]);
#if USE_TDM
    if (wave == 0) __builtin_amdgcn_s_wait_tensorcnt(0);
#endif
    __syncthreads();
  }

  // epilogue: C-frag row = v + 8*(lane>=16), col = lane%16
  const int rbase = m0 + wm + ((lane >> 4) << 3);
  const int cbase = n0 + wn + (lane & 15);
#pragma unroll
  for (int i = 0; i < 4; i++)
#pragma unroll
    for (int j = 0; j < 2; j++)
#pragma unroll
      for (int v = 0; v < 8; v++) {
        size_t r = (size_t)(rbase + i * 16 + v), c = (size_t)(cbase + j * 16);
        size_t o = r * (size_t)N + c;
        float val = acc[i][j][v];
        if (EPI == 0) {
          outb[o] = (bf16)val;
        } else if (EPI == 1) {
          outb[o] = (bf16)(val / (1.f + __expf(-val)));
        } else if (EPI == 2) {
          outb[o] = (bf16)(val * (float)mulm[o]);
        } else if (EPI == 3) {
          outf[o] = resid[o] + val;
        } else if (EPI == 4) {
          float al = 1.f / (1.f + __expf(-gatep[c]));
          outf[o] = resid[o] + al * val + (1.f - al) * (float)mixm[o];
        } else {                       // EPI==5: transposed store (for V^T)
          outb[c * (size_t)M + r] = (bf16)val;
        }
      }
}

// ---------------- flash attention ----------------
// block = (b, h, 128 q rows), 8 waves x 16 q rows. GQA: kv = h/4.
// Q stored [B*L, D] (col=h*64+hd); K stored [B*L, 256]; V stored TRANSPOSED [256, B*L].
__global__ __launch_bounds__(256) void attn_k(const bf16* __restrict__ Qg,
                                              const bf16* __restrict__ Kg,
                                              const bf16* __restrict__ Vt,
                                              const int* __restrict__ mask,
                                              bf16* __restrict__ Og) {
  __shared__ bf16 sK[2][64 * 72];     // [key][hd]   64 + 8 pad halfs
  __shared__ bf16 sV[2][64 * 72];     // [hd][key]   64 + 8 pad halfs
  __shared__ bf16 sP[8][16 * 72];     // per-wave P staging [qrow][key]
  const int tid = threadIdx.x, lane = tid & 31, wave = tid >> 5;
  const int b = blockIdx.z, h = blockIdx.y, qt = blockIdx.x;
  const int kv = h >> 2;
  const int kb8 = (lane >> 4) * 8, kb16 = (lane >> 4) * 16;

  const bf16* qp = Qg + (size_t)(b * L_ + qt * 128 + wave * 16 + (lane & 15)) * D_ + h * HD_;
  v16bf qa[2];
#pragma unroll
  for (int s = 0; s < 2; s++) qa[s] = frag_a(qp + s * 32 + kb8);

  v8f o[4]; v8f zero = {};
#pragma unroll
  for (int t = 0; t < 4; t++) o[t] = zero;
  float mrun[8], lrun[8];
#pragma unroll
  for (int v = 0; v < 8; v++) { mrun[v] = -1e30f; lrun[v] = 0.f; }

#if USE_TDM
  if (wave == 0) {                    // pad: 32 dwords data + 4 dwords -> 72-half rows
    tdm_load_2d(Kg + (size_t)(b * L_) * KVD_ + kv * HD_, lds_off(&sK[0][0]), 64, 64,
                (unsigned long long)KVD_, 1, 4, 3);
    tdm_load_2d(Vt + (size_t)(kv * HD_) * M_ + b * L_, lds_off(&sV[0][0]), 64, 64,
                (unsigned long long)M_, 1, 4, 3);
    __builtin_amdgcn_s_wait_tensorcnt(0);
  }
  __syncthreads();
#endif
  int buf = 0;
  for (int kc = 0; kc < L_; kc += 64, buf ^= 1) {
#if USE_TDM
    if (kc + 64 < L_ && wave == 0) {
      tdm_load_2d(Kg + (size_t)(b * L_ + kc + 64) * KVD_ + kv * HD_, lds_off(&sK[buf ^ 1][0]),
                  64, 64, (unsigned long long)KVD_, 1, 4, 3);
      tdm_load_2d(Vt + (size_t)(kv * HD_) * M_ + b * L_ + kc + 64, lds_off(&sV[buf ^ 1][0]),
                  64, 64, (unsigned long long)M_, 1, 4, 3);
    }
#else
    {   // synchronous fallback: both tiles row-contiguous 32B chunks
      int row = tid >> 2, seg = (tid & 3) * 16;
      const uint4* ks = (const uint4*)(Kg + (size_t)(b * L_ + kc + row) * KVD_ + kv * HD_ + seg);
      *(uint4*)&sK[buf][row * 72 + seg]     = ks[0];
      *(uint4*)&sK[buf][row * 72 + seg + 8] = ks[1];
      const uint4* vs = (const uint4*)(Vt + (size_t)(kv * HD_ + row) * M_ + b * L_ + kc + seg);
      *(uint4*)&sV[buf][row * 72 + seg]     = vs[0];
      *(uint4*)&sV[buf][row * 72 + seg + 8] = vs[1];
    }
    __syncthreads();
#endif
    // S = Q @ K^T over 4 key sub-tiles
    v8f S[4];
#pragma unroll
    for (int st = 0; st < 4; st++) {
      S[st] = zero;
#pragma unroll
      for (int s = 0; s < 2; s++) {
        v16bf kb = frag_b(&sK[buf][(st * 16 + (lane & 15)) * 72 + s * 32 + kb8]);
        S[st] = wmma_bf16(qa[s], kb, S[st]);
      }
    }
    // scale + mask + online softmax (rows in lane-halves -> shfl_xor < 16)
    float mnew[8];
#pragma unroll
    for (int v = 0; v < 8; v++) mnew[v] = -1e30f;
#pragma unroll
    for (int st = 0; st < 4; st++) {
      float madd = (mask[b * L_ + kc + st * 16 + (lane & 15)] == 0) ? -1e30f : 0.f;
#pragma unroll
      for (int v = 0; v < 8; v++) {
        S[st][v] = S[st][v] * 0.125f + madd;      // 1/sqrt(64)
        mnew[v] = fmaxf(mnew[v], S[st][v]);
      }
    }
#pragma unroll
    for (int v = 0; v < 8; v++) {
#pragma unroll
      for (int xo = 8; xo > 0; xo >>= 1) mnew[v] = fmaxf(mnew[v], __shfl_xor(mnew[v], xo, 32));
      float mn = fmaxf(mrun[v], mnew[v]);
      float rs = __expf(mrun[v] - mn);
      mrun[v] = mn; lrun[v] *= rs;
#pragma unroll
      for (int t = 0; t < 4; t++) o[t][v] *= rs;
    }
    float rsum[8];
#pragma unroll
    for (int v = 0; v < 8; v++) rsum[v] = 0.f;
#pragma unroll
    for (int st = 0; st < 4; st++)
#pragma unroll
      for (int v = 0; v < 8; v++) {
        float p = __expf(S[st][v] - mrun[v]);
        S[st][v] = p; rsum[v] += p;
      }
#pragma unroll
    for (int v = 0; v < 8; v++) {
#pragma unroll
      for (int xo = 8; xo > 0; xo >>= 1) rsum[v] += __shfl_xor(rsum[v], xo, 32);
      lrun[v] += rsum[v];
    }
    // P (C-layout) -> wave-private LDS -> A-fragments
#pragma unroll
    for (int st = 0; st < 4; st++)
#pragma unroll
      for (int v = 0; v < 8; v++)
        sP[wave][(v + ((lane >> 4) << 3)) * 72 + st * 16 + (lane & 15)] = (bf16)S[st][v];
    v16bf pa[2];
#pragma unroll
    for (int s = 0; s < 2; s++)
      pa[s] = frag_a(&sP[wave][(lane & 15) * 72 + s * 32 + kb8]);
    // O += P @ V
#pragma unroll
    for (int t = 0; t < 4; t++)
#pragma unroll
      for (int s = 0; s < 2; s++) {
        v16bf vb = frag_b(&sV[buf][(t * 16 + (lane & 15)) * 72 + s * 32 + kb8]);
        o[t] = wmma_bf16(pa[s], vb, o[t]);
      }
#if USE_TDM
    if (wave == 0) __builtin_amdgcn_s_wait_tensorcnt(0);
#endif
    __syncthreads();
  }
  // write O / l  (attn layout [B*L, D], col = h*64 + hd)
#pragma unroll
  for (int t = 0; t < 4; t++)
#pragma unroll
    for (int v = 0; v < 8; v++) {
      int r = qt * 128 + wave * 16 + v + ((lane >> 4) << 3);
      int c = h * HD_ + t * 16 + (lane & 15);
      Og[(size_t)(b * L_ + r) * D_ + c] = (bf16)(o[t][v] / lrun[v]);
    }
}

// ---------------- launcher ----------------
extern "C" void kernel_launch(void* const* d_in, const int* in_sizes, int n_in,
                              void* d_out, int out_size, void* d_ws, size_t ws_size,
                              hipStream_t stream) {
  const float* x     = (const float*)d_in[0];
  const int*   mask  = (const int*)d_in[1];
  const float* Wq    = (const float*)d_in[2];
  const float* Wk    = (const float*)d_in[3];
  const float* Wv    = (const float*)d_in[4];
  const float* Wo    = (const float*)d_in[5];
  const float* ln1g  = (const float*)d_in[6];
  const float* ln1b  = (const float*)d_in[7];
  const float* ln2g  = (const float*)d_in[8];
  const float* ln2b  = (const float*)d_in[9];
  const float* Wg    = (const float*)d_in[10];
  const float* Wu    = (const float*)d_in[11];
  const float* Wd    = (const float*)d_in[12];
  const float* gparm = (const float*)d_in[13];
  float* outp = (float*)d_out;

  char* ws = (char*)d_ws; size_t off = 0;
  auto alloc = [&](size_t bytes) -> void* {
    void* p = ws + off; off = (off + bytes + 255) & ~(size_t)255; return p;
  };
  bf16* xn   = (bf16*)alloc((size_t)M_ * D_ * 2);     // xn1, reused as xn2
  bf16* wqT  = (bf16*)alloc((size_t)D_ * D_ * 2);     // all weights stored [N,K]
  bf16* wkT  = (bf16*)alloc((size_t)D_ * KVD_ * 2);
  bf16* wvT  = (bf16*)alloc((size_t)D_ * KVD_ * 2);
  bf16* woT  = (bf16*)alloc((size_t)D_ * D_ * 2);
  bf16* wgT  = (bf16*)alloc((size_t)D_ * FF_ * 2);
  bf16* wuT  = (bf16*)alloc((size_t)D_ * FF_ * 2);
  bf16* wdT  = (bf16*)alloc((size_t)FF_ * D_ * 2);
  bf16* qB   = (bf16*)alloc((size_t)M_ * D_ * 2);
  bf16* kB   = (bf16*)alloc((size_t)M_ * KVD_ * 2);   // [B*L, 256]
  bf16* vT   = (bf16*)alloc((size_t)M_ * KVD_ * 2);   // transposed [256, B*L]
  bf16* attn = (bf16*)alloc((size_t)M_ * D_ * 2);
  float* x2  = (float*)alloc((size_t)M_ * D_ * 4);
  bf16* gate = (bf16*)alloc((size_t)M_ * FF_ * 2);    // gate, then h=gate*up

  auto cvtT = [&](const float* src, bf16* dst, int K, int N) {
    cvt_transpose_k<<<dim3(N / 32, K / 32), 256, 0, stream>>>(src, dst, K, N);
  };
  // LN1
  layernorm_bf16_k<<<M_, 256, 0, stream>>>(x, ln1g, ln1b, xn);
  // weight convert+transpose (once per call)
  cvtT(Wq, wqT, D_, D_);     cvtT(Wk, wkT, D_, KVD_);
  cvtT(Wv, wvT, D_, KVD_);   cvtT(Wo, woT, D_, D_);
  cvtT(Wg, wgT, D_, FF_);    cvtT(Wu, wuT, D_, FF_);
  cvtT(Wd, wdT, FF_, D_);
  // QKV projections (V stored transposed for attention)
  gemm_bf16_k<0><<<dim3(D_ / 128, M_ / 128), 256, 0, stream>>>(
      xn, wqT, M_, D_, D_, nullptr, nullptr, nullptr, nullptr, nullptr, qB);
  gemm_bf16_k<0><<<dim3(KVD_ / 128, M_ / 128), 256, 0, stream>>>(
      xn, wkT, M_, KVD_, D_, nullptr, nullptr, nullptr, nullptr, nullptr, kB);
  gemm_bf16_k<5><<<dim3(KVD_ / 128, M_ / 128), 256, 0, stream>>>(
      xn, wvT, M_, KVD_, D_, nullptr, nullptr, nullptr, nullptr, nullptr, vT);
  // attention
  attn_k<<<dim3(L_ / 128, H_, B_), 256, 0, stream>>>(qB, kB, vT, mask, attn);
  // x2 = x + attn @ Wo
  gemm_bf16_k<3><<<dim3(D_ / 128, M_ / 128), 256, 0, stream>>>(
      attn, woT, M_, D_, D_, x, nullptr, nullptr, nullptr, x2, nullptr);
  // LN2 (reuse xn)
  layernorm_bf16_k<<<M_, 256, 0, stream>>>(x2, ln2g, ln2b, xn);
  // gate = silu(xn @ Wgate)
  gemm_bf16_k<1><<<dim3(FF_ / 128, M_ / 128), 256, 0, stream>>>(
      xn, wgT, M_, FF_, D_, nullptr, nullptr, nullptr, nullptr, nullptr, gate);
  // h = (xn @ Wup) * gate  (in-place over gate buffer)
  gemm_bf16_k<2><<<dim3(FF_ / 128, M_ / 128), 256, 0, stream>>>(
      xn, wuT, M_, FF_, D_, nullptr, gate, nullptr, nullptr, nullptr, gate);
  // out = x2 + alpha*(h @ Wdown) + (1-alpha)*xn
  gemm_bf16_k<4><<<dim3(D_ / 128, M_ / 128), 256, 0, stream>>>(
      gate, wdT, M_, D_, FF_, x2, nullptr, xn, gparm, outp, nullptr);
  (void)in_sizes; (void)n_in; (void)out_size; (void)ws_size;
}
